// AdaptiveAttentionHead_16784732192930
// MI455X (gfx1250) — matrix operand order repaired and checked by reading, not compiled, not run on hardware
//
#include <hip/hip_runtime.h>
#include <hip/hip_bf16.h>

typedef __attribute__((ext_vector_type(16))) _Float16     v16h;
typedef __attribute__((ext_vector_type(8)))  float        v8f;
typedef __attribute__((ext_vector_type(8)))  unsigned int v8u;

#define N_EMBD 1024
#define HEADSZ 64
#define NPROJ  192   // q|k|v concatenated columns
#define TLEN   2048
#define BATCH  4
#define LDP    40    // proj LDS stride (32+8 halfs)
#define LDA    72    // attn LDS stride (64+8 halfs); 144B row = 16B-aligned

// ---- CDNA5 async global->LDS (ASYNCcnt) ----
__device__ inline void async_ld16(unsigned ldsOff, const void* g) {
  asm volatile("global_load_async_to_lds_b128 %0, %1, off"
               :: "v"(ldsOff), "v"(g) : "memory");
}
__device__ inline void wait_async0() {
  asm volatile("s_wait_asynccnt 0x0" ::: "memory");
}

// ---- WMMA fragment gathers (layouts per CDNA5 ISA 7.12.2, wave32) ----

// A 16x32 f16: row = rowBase + lane%16.
// lanes 0-15:  VGPR0-3 -> K 0..7,  VGPR4-7 -> K 16..23
// lanes 16-31: VGPR0-3 -> K 8..15, VGPR4-7 -> K 24..31
__device__ inline v16h frag_a(const _Float16* lds, int rowBase, int stride, int kBase) {
  const int lane = threadIdx.x & 31;
  const int row  = rowBase + (lane & 15);
  const int kh   = (lane < 16) ? 0 : 8;
  const unsigned int* p = (const unsigned int*)(lds + row * stride);
  union { v8u u; v16h h; } r;
#pragma unroll
  for (int v = 0; v < 8; ++v) {
    int k = ((v < 4) ? 0 : 16) + kh + 2 * (v & 3);
    r.u[v] = p[(kBase + k) >> 1];
  }
  return r.h;
}

// B 32x16 f16 from LDS laid out [col][k] (pairs contiguous along matmul-K):
// col = colBase + lane%16; lanes 0-15: K 0..7,16..23; lanes 16-31: K 8..15,24..31 (from kBase)
__device__ inline v16h frag_b(const _Float16* lds, int colBase, int stride, int kBase) {
  const int lane = threadIdx.x & 31;
  const int col  = colBase + (lane & 15);
  const int kb   = kBase + ((lane < 16) ? 0 : 8);
  const unsigned int* p = (const unsigned int*)(lds + col * stride);
  union { v8u u; v16h h; } r;
#pragma unroll
  for (int v = 0; v < 8; ++v) r.u[v] = p[(kb + ((v < 4) ? 0 : 16) + 2 * (v & 3)) >> 1];
  return r.h;
}

// ---- Kernel 0: fold SVF factors: Wt[col][c] = sum_r U[h][r]*z[r]*V[r][c] ----
__global__ __launch_bounds__(256) void svf_combine_kernel(
    const float* __restrict__ Vq, const float* __restrict__ Uq, const float* __restrict__ zq,
    const float* __restrict__ Vk, const float* __restrict__ Uk, const float* __restrict__ zk,
    const float* __restrict__ Vv, const float* __restrict__ Uv, const float* __restrict__ zv,
    _Float16* __restrict__ Wt) {
  int idx = blockIdx.x * 256 + threadIdx.x;
  if (idx >= NPROJ * N_EMBD) return;
  int c   = idx & (N_EMBD - 1);
  int col = idx >> 10;       // 0..191
  int n   = col >> 6;        // 0=q 1=k 2=v
  int h   = col & 63;
  const float* V = (n == 0) ? Vq : (n == 1) ? Vk : Vv;
  const float* U = (n == 0) ? Uq : (n == 1) ? Uk : Uv;
  const float* z = (n == 0) ? zq : (n == 1) ? zk : zv;
  float s = 0.f;
#pragma unroll
  for (int r = 0; r < 16; ++r) s += U[h * 16 + r] * z[r] * V[r * N_EMBD + c];
  Wt[idx] = (_Float16)s;
}

// ---- Kernel 1: qkv[8192][192] = x[8192][1024] * Wt^T, f16 out, scale q by 1/8 ----
__global__ __launch_bounds__(256) void svf_proj_kernel(
    const float* __restrict__ xg, const _Float16* __restrict__ Wt,
    _Float16* __restrict__ qkv) {
  __shared__ __align__(16) _Float16 As[128 * LDP];
  __shared__ __align__(16) _Float16 Bs[64 * LDP];
  const int t = threadIdx.x, lane = t & 31, wave = t >> 5;
  const int wr = wave & 3, wc = wave >> 2;           // 4x2 wave grid, 32x32 per wave
  const int mBase = blockIdx.x * 128, nBase = blockIdx.y * 64;
  v8f acc[2][2] = {};
  for (int k0 = 0; k0 < N_EMBD; k0 += 32) {
    { // A tile: 128x32 f32 -> f16 LDS
      int row = t >> 1, ch = (t & 1) * 16;
      const float4* gp = (const float4*)(xg + (size_t)(mBase + row) * N_EMBD + k0 + ch);
      unsigned int* lp = (unsigned int*)(As + row * LDP + ch);
#pragma unroll
      for (int i = 0; i < 4; ++i) {
        float4 f = gp[i];
        union { unsigned int u[2]; _Float16 h[4]; } pk;
        pk.h[0] = (_Float16)f.x; pk.h[1] = (_Float16)f.y;
        pk.h[2] = (_Float16)f.z; pk.h[3] = (_Float16)f.w;
        lp[2 * i] = pk.u[0]; lp[2 * i + 1] = pk.u[1];
      }
    }
    { // B tile: Wt rows nBase..+63, cols k0..+31, stored [col][k]
      int col = t >> 2, ch = (t & 3) * 8;
      uint4 d = *(const uint4*)(Wt + (size_t)(nBase + col) * N_EMBD + k0 + ch);
      *(uint4*)(Bs + col * LDP + ch) = d;     // col*80B is 16B aligned
    }
    __syncthreads();
    v16h a0 = frag_a(As, wr * 32,      LDP, 0);
    v16h a1 = frag_a(As, wr * 32 + 16, LDP, 0);
    v16h b0 = frag_b(Bs, wc * 32,      LDP, 0);
    v16h b1 = frag_b(Bs, wc * 32 + 16, LDP, 0);
    acc[0][0] = __builtin_amdgcn_wmma_f32_16x16x32_f16(false, a0, false, b0, (short)0, acc[0][0], false, false);
    acc[0][1] = __builtin_amdgcn_wmma_f32_16x16x32_f16(false, a0, false, b1, (short)0, acc[0][1], false, false);
    acc[1][0] = __builtin_amdgcn_wmma_f32_16x16x32_f16(false, a1, false, b0, (short)0, acc[1][0], false, false);
    acc[1][1] = __builtin_amdgcn_wmma_f32_16x16x32_f16(false, a1, false, b1, (short)0, acc[1][1], false, false);
    __syncthreads();
  }
  const int ln = lane & 15, hi = (lane >> 4) * 8;
#pragma unroll
  for (int i = 0; i < 2; ++i)
#pragma unroll
    for (int jj = 0; jj < 2; ++jj)
#pragma unroll
      for (int v = 0; v < 8; ++v) {
        int row = mBase + wr * 32 + i * 16 + v + hi;
        int col = nBase + wc * 32 + jj * 16 + ln;
        float val = acc[i][jj][v];
        if (col < HEADSZ) val *= 0.125f;          // fold q scale = 1/sqrt(64)
        qkv[(size_t)row * NPROJ + col] = (_Float16)val;
      }
}

// ---- Kernel 2: causal flash attention, Br=64, Bc=64, 4 waves x 16 rows,
// ----           double-buffered: K via async global->LDS, V via VGPR pipeline
__global__ __launch_bounds__(128) void flash_attn_kernel(
    const _Float16* __restrict__ qkv, float* __restrict__ out) {
  __shared__ __align__(16) _Float16 Qs[64 * LDA];
  __shared__ __align__(16) _Float16 Ks[2][64 * LDA];
  __shared__ __align__(16) _Float16 Vt[2][64 * LDA];   // transposed: [h][n]
  __shared__ __align__(16) _Float16 Ps[4 * 16 * LDA];  // per-wave P staging
  const int t = threadIdx.x, lane = t & 31, wave = t >> 5;
  const int ln = lane & 15, hi = (lane >> 4) * 8;
  const int qs = blockIdx.x * 64;
  const size_t rowOff = (size_t)blockIdx.y * TLEN;
  const int nBlk = (qs >> 6) + 1;
  const int vn = t >> 1, vc = (t & 1) * 32;   // cooperative row / 32-half chunk

  // Q tile (cols 0..63 of qkv rows)
  {
    const uint4* gp = (const uint4*)(qkv + (rowOff + qs + vn) * NPROJ + vc);
    unsigned int* lp = (unsigned int*)(Qs + vn * LDA + vc);
#pragma unroll
    for (int i = 0; i < 4; ++i) {
      uint4 d = gp[i];
      lp[4 * i] = d.x; lp[4 * i + 1] = d.y; lp[4 * i + 2] = d.z; lp[4 * i + 3] = d.w;
    }
  }

  uint4 vreg[4];  // V prefetch registers
  auto issueK = [&](int kb, int buf) {      // async global->LDS, row-major [n][d]
    const _Float16* g = qkv + (rowOff + kb + vn) * NPROJ + HEADSZ + vc;
    unsigned l0 = (unsigned)(uintptr_t)(&Ks[buf][vn * LDA + vc]);
#pragma unroll
    for (int i = 0; i < 4; ++i) async_ld16(l0 + i * 16, (const void*)(g + i * 8));
  };
  auto loadV = [&](int kb) {                // start global loads into VGPRs
    const uint4* gp = (const uint4*)(qkv + (rowOff + kb + vn) * NPROJ + 2 * HEADSZ + vc);
#pragma unroll
    for (int i = 0; i < 4; ++i) vreg[i] = gp[i];
  };
  auto storeV = [&](int buf) {              // transpose-store: Vt[h][n]
#pragma unroll
    for (int i = 0; i < 4; ++i) {
      union { uint4 u; _Float16 h[8]; } d; d.u = vreg[i];
#pragma unroll
      for (int k = 0; k < 8; ++k) Vt[buf][(vc + i * 8 + k) * LDA + vn] = d.h[k];
    }
  };

  issueK(0, 0);
  loadV(0);
  storeV(0);
  wait_async0();
  __syncthreads();

  const v16h aq0 = frag_a(Qs, wave * 16, LDA, 0);
  const v16h aq1 = frag_a(Qs, wave * 16, LDA, 32);

  v8f O[4] = {};
  float m[8], l[8];
#pragma unroll
  for (int v = 0; v < 8; ++v) { m[v] = -3.0e38f; l[v] = 0.f; }
  const int rowBaseW = qs + wave * 16;

  for (int j = 0; j < nBlk; ++j) {
    const int cur = j & 1, nxt = cur ^ 1;
    const int kb = j * 64;
    const bool pf = (j + 1 < nBlk);
    if (pf) { issueK((j + 1) * 64, nxt); loadV((j + 1) * 64); }  // overlap next tile
    const _Float16* Kc = Ks[cur];
    const _Float16* Vc = Vt[cur];

    // S = Q * K^T (f32 accum), scale pre-folded into q
    v8f S[4];
#pragma unroll
    for (int nt = 0; nt < 4; ++nt) {
      v8f z = {};
      z = __builtin_amdgcn_wmma_f32_16x16x32_f16(false, aq0, false,
            frag_b(Kc, nt * 16, LDA, 0),  (short)0, z, false, false);
      z = __builtin_amdgcn_wmma_f32_16x16x32_f16(false, aq1, false,
            frag_b(Kc, nt * 16, LDA, 32), (short)0, z, false, false);
      S[nt] = z;
    }
    if (kb + 63 > rowBaseW) { // causal mask only on diagonal-crossing blocks
#pragma unroll
      for (int nt = 0; nt < 4; ++nt) {
        int colg = kb + nt * 16 + ln;
#pragma unroll
        for (int v = 0; v < 8; ++v)
          if (colg > rowBaseW + v + hi) S[nt][v] = -3.0e38f;
      }
    }
    // online softmax: row stats via xor-butterfly over 16-lane halves
    _Float16* Pw = Ps + wave * 16 * LDA;
#pragma unroll
    for (int v = 0; v < 8; ++v) {
      float rm = fmaxf(fmaxf(S[0][v], S[1][v]), fmaxf(S[2][v], S[3][v]));
      rm = fmaxf(rm, __shfl_xor(rm, 1, 32));
      rm = fmaxf(rm, __shfl_xor(rm, 2, 32));
      rm = fmaxf(rm, __shfl_xor(rm, 4, 32));
      rm = fmaxf(rm, __shfl_xor(rm, 8, 32));
      float mn = fmaxf(m[v], rm);
      float corr = __expf(m[v] - mn);
      m[v] = mn;
      float rs = 0.f;
#pragma unroll
      for (int nt = 0; nt < 4; ++nt) {
        float p = __expf(S[nt][v] - mn);
        rs += p;
        Pw[(v + hi) * LDA + nt * 16 + ln] = (_Float16)p;
      }
      rs += __shfl_xor(rs, 1, 32);
      rs += __shfl_xor(rs, 2, 32);
      rs += __shfl_xor(rs, 4, 32);
      rs += __shfl_xor(rs, 8, 32);
      l[v] = l[v] * corr + rs;
#pragma unroll
      for (int ht = 0; ht < 4; ++ht) O[ht][v] *= corr;
    }
    // per-wave LDS staging: DS pipeline is in-order within a wave; pin it
    asm volatile("s_wait_dscnt 0x0" ::: "memory");
    v16h ap0 = frag_a(Pw, 0, LDA, 0);
    v16h ap1 = frag_a(Pw, 0, LDA, 32);
#pragma unroll
    for (int ht = 0; ht < 4; ++ht) {
      O[ht] = __builtin_amdgcn_wmma_f32_16x16x32_f16(false, ap0, false,
                frag_b(Vc, ht * 16, LDA, 0),  (short)0, O[ht], false, false);
      O[ht] = __builtin_amdgcn_wmma_f32_16x16x32_f16(false, ap1, false,
                frag_b(Vc, ht * 16, LDA, 32), (short)0, O[ht], false, false);
    }
    if (pf) storeV(nxt);   // waits its own loadcnt; lands in the other buffer
    wait_async0();         // next K tile resident in LDS
    __syncthreads();       // all waves done reading cur / writing nxt
  }
#pragma unroll
  for (int v = 0; v < 8; ++v) {
    float inv = 1.0f / l[v];
    size_t row = rowOff + rowBaseW + v + hi;
#pragma unroll
    for (int ht = 0; ht < 4; ++ht)
      out[row * HEADSZ + ht * 16 + ln] = O[ht][v] * inv;
  }
}

extern "C" void kernel_launch(void* const* d_in, const int* in_sizes, int n_in,
                              void* d_out, int out_size, void* d_ws, size_t ws_size,
                              hipStream_t stream) {
  const float* x  = (const float*)d_in[0];
  const float* Vq = (const float*)d_in[1];
  const float* Uq = (const float*)d_in[2];
  const float* zq = (const float*)d_in[3];
  const float* Vk = (const float*)d_in[4];
  const float* Uk = (const float*)d_in[5];
  const float* zk = (const float*)d_in[6];
  const float* Vv = (const float*)d_in[7];
  const float* Uv = (const float*)d_in[8];
  const float* zv = (const float*)d_in[9];

  _Float16* Wt  = (_Float16*)d_ws;                                       // [192][1024] f16
  _Float16* qkv = (_Float16*)((char*)d_ws + (size_t)NPROJ * N_EMBD * 2); // [8192][192] f16

  svf_combine_kernel<<<(NPROJ * N_EMBD + 255) / 256, 256, 0, stream>>>(
      Vq, Uq, zq, Vk, Uk, zk, Vv, Uv, zv, Wt);
  svf_proj_kernel<<<dim3((BATCH * TLEN) / 128, NPROJ / 64), 256, 0, stream>>>(x, Wt, qkv);
  flash_attn_kernel<<<dim3(TLEN / 64, BATCH), 128, 0, stream>>>(qkv, (float*)d_out);
}
